// Seq2seq_35064113005057
// MI455X (gfx1250) — matrix operand order repaired and checked by reading, not compiled, not run on hardware
//
#include <hip/hip_runtime.h>

// Seq2seq GRU + local attention for MI455X (gfx1250), bf16 WMMA everywhere.
// Workspace layout needs ~266 MB (enc_w_full fp32 131MB + enc_out bf16 65MB +
// src bf16 33MB + bf16 weights ~30MB + small ping-pong state buffers).

#define B_    64
#define LSRC  500
#define TTRG  64
#define DIMM  512
#define HH    1024
#define DD    6
#define WINN  13

typedef __bf16 bf16;
typedef __attribute__((ext_vector_type(16))) __bf16 v16bf;
typedef __attribute__((ext_vector_type(8)))  __bf16 v8bf;
typedef __attribute__((ext_vector_type(8)))  float  v8f;

__device__ __forceinline__ float sigmoidf_(float x) { return 1.0f / (1.0f + __expf(-x)); }

__device__ __forceinline__ v8f wmma_bf16(v16bf a, v16bf b, v8f c) {
  // (neg_a, A, neg_b, B, c_mod, C, reuse_a, reuse_b)
  return __builtin_amdgcn_wmma_f32_16x16x32_bf16(false, a, false, b, (short)0, c, false, false);
}

// A fragment: 16x32 bf16 tile of row-major A (row stride ld elements), tile at (m0,k0).
// Layout used here: lane m = lane&15, half hi = lane>>4; elems 0..15 = A[m, k0+hi*16 .. +15]
// (same contiguous k order as load_b_frag).  Two 16B contiguous loads per lane.
__device__ __forceinline__ v16bf load_a_frag(const bf16* __restrict__ A, long ld, int m0, int k0) {
  const int lane = threadIdx.x & 31;
  const bf16* p = A + (long)(m0 + (lane & 15)) * ld + k0 + ((lane >> 4) << 4);
  v8bf lo = *(const v8bf*)p;
  v8bf hi = *(const v8bf*)(p + 8);
  return __builtin_shufflevector(lo, hi, 0,1,2,3,4,5,6,7,8,9,10,11,12,13,14,15);
}

// B fragment: 32x16 (KxN) bf16 tile where the B matrix is W^T with W stored row-major
// [N, K] (torch weight layout). Lane n = lane&15, K range = k0 + (lane>>4)*16 .. +15:
// 32 contiguous bytes per lane straight out of W's row n.
__device__ __forceinline__ v16bf load_b_frag(const bf16* __restrict__ W, long ld, int n0, int k0) {
  const int lane = threadIdx.x & 31;
  return *(const v16bf*)(W + (long)(n0 + (lane & 15)) * ld + k0 + ((lane >> 4) << 4));
}

// ---------------------------------------------------------------------------
// Conversion / init kernels
// ---------------------------------------------------------------------------
__global__ void cvt_bf16_kernel(const float* __restrict__ s, bf16* __restrict__ d, long n) {
  long i = (long)blockIdx.x * blockDim.x + threadIdx.x;
  const long st = (long)gridDim.x * blockDim.x;
  for (; i < n; i += st) d[i] = (bf16)s[i];
}

__global__ void zero_f32_kernel(float* __restrict__ p, long n) {
  long i = (long)blockIdx.x * blockDim.x + threadIdx.x;
  const long st = (long)gridDim.x * blockDim.x;
  for (; i < n; i += st) p[i] = 0.0f;
}

// dec_inputs[:,0]=0 ; dec_inputs[:,t]=trg[:,t-1]  (teacher forcing, converted to bf16)
__global__ void shift_trg_kernel(const float* __restrict__ trg, bf16* __restrict__ dx) {
  const long n = (long)B_ * TTRG * DIMM;
  long i = (long)blockIdx.x * blockDim.x + threadIdx.x;
  const long st = (long)gridDim.x * blockDim.x;
  for (; i < n; i += st) {
    const int j = (int)(i % DIMM);
    const long r = i / DIMM;
    const int t = (int)(r % TTRG);
    const long b = r / TTRG;
    dx[i] = (t == 0) ? (bf16)0.0f : (bf16)trg[((long)b * TTRG + (t - 1)) * DIMM + j];
  }
}

// ---------------------------------------------------------------------------
// Big parallel GEMM:  C[M,N] = A_bf16[M,K] @ W_bf16[N,K]^T + bias   (fp32 out)
// block = 64x64 tile, 8 waves, 2 accum tiles per wave.
// ---------------------------------------------------------------------------
__global__ __launch_bounds__(256) void gemm_bias_kernel(
    const bf16* __restrict__ A, const bf16* __restrict__ W,
    const float* __restrict__ bias, float* __restrict__ C, int N, int K)
{
  const int n0 = blockIdx.x * 64, m0 = blockIdx.y * 64;
  const int wave = threadIdx.x >> 5, lane = threadIdx.x & 31;
  const int ni = wave & 3, mi = wave >> 2;
  const int nt = n0 + ni * 16;
  v8f acc0 = {}, acc1 = {};
#pragma unroll 2
  for (int k0 = 0; k0 < K; k0 += 32) {
    v16bf b  = load_b_frag(W, K, nt, k0);
    v16bf a0 = load_a_frag(A, K, m0 + mi * 16, k0);
    v16bf a1 = load_a_frag(A, K, m0 + (mi + 2) * 16, k0);
    acc0 = wmma_bf16(a0, b, acc0);
    acc1 = wmma_bf16(a1, b, acc1);
  }
  const int n = nt + (lane & 15);
  const float bb = bias[n];
  const int ma = m0 + mi * 16 + (lane >> 4) * 8;
  const int mb = m0 + (mi + 2) * 16 + (lane >> 4) * 8;
#pragma unroll
  for (int v = 0; v < 8; ++v) {
    C[(long)(ma + v) * N + n] = acc0[v] + bb;
    C[(long)(mb + v) * N + n] = acc1[v] + bb;
  }
}

// ---------------------------------------------------------------------------
// Encoder GRU step (fused x-GEMM + h-GEMM + gates).  grid = 16 blocks, each
// owns h-columns [n0,n0+64) so all three gates fuse locally (no inter-WG sync).
// ---------------------------------------------------------------------------
__global__ __launch_bounds__(256) void enc_step_kernel(
    const bf16* __restrict__ x, long ldx,
    const bf16* __restrict__ hbf, const float* __restrict__ hf,
    const bf16* __restrict__ Wih, const bf16* __restrict__ Whh,
    const float* __restrict__ bih, const float* __restrict__ bhh,
    float* __restrict__ hof, bf16* __restrict__ hob,
    bf16* __restrict__ eob, long ldeo)
{
  const int n0 = blockIdx.x * 64;
  const int wave = threadIdx.x >> 5, lane = threadIdx.x & 31;
  const int ni = wave & 3, mi = wave >> 2;
  const int nt = n0 + ni * 16;
  v8f ar0 = {}, ar1 = {}, az0 = {}, az1 = {}, ai0 = {}, ai1 = {}, ah0 = {}, ah1 = {};
#pragma unroll 2
  for (int k0 = 0; k0 < DIMM; k0 += 32) {          // input side (x @ Wih^T)
    v16bf a0 = load_a_frag(x, ldx, mi * 16, k0);
    v16bf a1 = load_a_frag(x, ldx, (mi + 2) * 16, k0);
    v16bf br = load_b_frag(Wih, DIMM, 0 * HH + nt, k0);
    v16bf bz = load_b_frag(Wih, DIMM, 1 * HH + nt, k0);
    v16bf bn = load_b_frag(Wih, DIMM, 2 * HH + nt, k0);
    ar0 = wmma_bf16(a0, br, ar0); ar1 = wmma_bf16(a1, br, ar1);
    az0 = wmma_bf16(a0, bz, az0); az1 = wmma_bf16(a1, bz, az1);
    ai0 = wmma_bf16(a0, bn, ai0); ai1 = wmma_bf16(a1, bn, ai1);
  }
#pragma unroll 2
  for (int k0 = 0; k0 < HH; k0 += 32) {            // hidden side (h @ Whh^T)
    v16bf a0 = load_a_frag(hbf, HH, mi * 16, k0);
    v16bf a1 = load_a_frag(hbf, HH, (mi + 2) * 16, k0);
    v16bf br = load_b_frag(Whh, HH, 0 * HH + nt, k0);
    v16bf bz = load_b_frag(Whh, HH, 1 * HH + nt, k0);
    v16bf bn = load_b_frag(Whh, HH, 2 * HH + nt, k0);
    ar0 = wmma_bf16(a0, br, ar0); ar1 = wmma_bf16(a1, br, ar1);
    az0 = wmma_bf16(a0, bz, az0); az1 = wmma_bf16(a1, bz, az1);
    ah0 = wmma_bf16(a0, bn, ah0); ah1 = wmma_bf16(a1, bn, ah1);  // n-gate hidden kept separate
  }
  const int n = nt + (lane & 15);
  const float brb = bih[n] + bhh[n];
  const float bzb = bih[HH + n] + bhh[HH + n];
  const float bni = bih[2 * HH + n];
  const float bnh = bhh[2 * HH + n];
  const int mb0 = mi * 16 + (lane >> 4) * 8;
  const int mb1 = (mi + 2) * 16 + (lane >> 4) * 8;
#pragma unroll
  for (int v = 0; v < 8; ++v) {
    {
      const int m = mb0 + v;
      const float r = sigmoidf_(ar0[v] + brb);
      const float z = sigmoidf_(az0[v] + bzb);
      const float nn = tanhf(ai0[v] + bni + r * (ah0[v] + bnh));
      const float hn = (1.0f - z) * nn + z * hf[(long)m * HH + n];
      hof[(long)m * HH + n] = hn;
      const bf16 hb = (bf16)hn;
      hob[(long)m * HH + n] = hb;
      eob[(long)m * ldeo + n] = hb;
    }
    {
      const int m = mb1 + v;
      const float r = sigmoidf_(ar1[v] + brb);
      const float z = sigmoidf_(az1[v] + bzb);
      const float nn = tanhf(ai1[v] + bni + r * (ah1[v] + bnh));
      const float hn = (1.0f - z) * nn + z * hf[(long)m * HH + n];
      hof[(long)m * HH + n] = hn;
      const bf16 hb = (bf16)hn;
      hob[(long)m * HH + n] = hb;
      eob[(long)m * ldeo + n] = hb;
    }
  }
}

// ---------------------------------------------------------------------------
// Decoder GRU step: rnn_in = [x_t, c] -> K loops over x (Wih cols 0..511),
// c (Wih cols 512..1535) and h (Whh).  Same gate split.
// ---------------------------------------------------------------------------
__global__ __launch_bounds__(256) void dec_step_kernel(
    const bf16* __restrict__ x, long ldx,
    const bf16* __restrict__ cbf,
    const bf16* __restrict__ hbf, const float* __restrict__ hf,
    const bf16* __restrict__ Wih, const bf16* __restrict__ Whh,
    const float* __restrict__ bih, const float* __restrict__ bhh,
    float* __restrict__ hof, bf16* __restrict__ hob)
{
  const int n0 = blockIdx.x * 64;
  const int wave = threadIdx.x >> 5, lane = threadIdx.x & 31;
  const int ni = wave & 3, mi = wave >> 2;
  const int nt = n0 + ni * 16;
  const long ldW = DIMM + HH;  // 1536
  v8f ar0 = {}, ar1 = {}, az0 = {}, az1 = {}, ai0 = {}, ai1 = {}, ah0 = {}, ah1 = {};
#pragma unroll 2
  for (int k0 = 0; k0 < DIMM; k0 += 32) {          // x part of input side
    v16bf a0 = load_a_frag(x, ldx, mi * 16, k0);
    v16bf a1 = load_a_frag(x, ldx, (mi + 2) * 16, k0);
    v16bf br = load_b_frag(Wih, ldW, 0 * HH + nt, k0);
    v16bf bz = load_b_frag(Wih, ldW, 1 * HH + nt, k0);
    v16bf bn = load_b_frag(Wih, ldW, 2 * HH + nt, k0);
    ar0 = wmma_bf16(a0, br, ar0); ar1 = wmma_bf16(a1, br, ar1);
    az0 = wmma_bf16(a0, bz, az0); az1 = wmma_bf16(a1, bz, az1);
    ai0 = wmma_bf16(a0, bn, ai0); ai1 = wmma_bf16(a1, bn, ai1);
  }
#pragma unroll 2
  for (int k0 = 0; k0 < HH; k0 += 32) {            // context part of input side
    v16bf a0 = load_a_frag(cbf, HH, mi * 16, k0);
    v16bf a1 = load_a_frag(cbf, HH, (mi + 2) * 16, k0);
    v16bf br = load_b_frag(Wih, ldW, 0 * HH + nt, DIMM + k0);
    v16bf bz = load_b_frag(Wih, ldW, 1 * HH + nt, DIMM + k0);
    v16bf bn = load_b_frag(Wih, ldW, 2 * HH + nt, DIMM + k0);
    ar0 = wmma_bf16(a0, br, ar0); ar1 = wmma_bf16(a1, br, ar1);
    az0 = wmma_bf16(a0, bz, az0); az1 = wmma_bf16(a1, bz, az1);
    ai0 = wmma_bf16(a0, bn, ai0); ai1 = wmma_bf16(a1, bn, ai1);
  }
#pragma unroll 2
  for (int k0 = 0; k0 < HH; k0 += 32) {            // hidden side
    v16bf a0 = load_a_frag(hbf, HH, mi * 16, k0);
    v16bf a1 = load_a_frag(hbf, HH, (mi + 2) * 16, k0);
    v16bf br = load_b_frag(Whh, HH, 0 * HH + nt, k0);
    v16bf bz = load_b_frag(Whh, HH, 1 * HH + nt, k0);
    v16bf bn = load_b_frag(Whh, HH, 2 * HH + nt, k0);
    ar0 = wmma_bf16(a0, br, ar0); ar1 = wmma_bf16(a1, br, ar1);
    az0 = wmma_bf16(a0, bz, az0); az1 = wmma_bf16(a1, bz, az1);
    ah0 = wmma_bf16(a0, bn, ah0); ah1 = wmma_bf16(a1, bn, ah1);
  }
  const int n = nt + (lane & 15);
  const float brb = bih[n] + bhh[n];
  const float bzb = bih[HH + n] + bhh[HH + n];
  const float bni = bih[2 * HH + n];
  const float bnh = bhh[2 * HH + n];
  const int mb0 = mi * 16 + (lane >> 4) * 8;
  const int mb1 = (mi + 2) * 16 + (lane >> 4) * 8;
#pragma unroll
  for (int v = 0; v < 8; ++v) {
    {
      const int m = mb0 + v;
      const float r = sigmoidf_(ar0[v] + brb);
      const float z = sigmoidf_(az0[v] + bzb);
      const float nn = tanhf(ai0[v] + bni + r * (ah0[v] + bnh));
      const float hn = (1.0f - z) * nn + z * hf[(long)m * HH + n];
      hof[(long)m * HH + n] = hn;
      hob[(long)m * HH + n] = (bf16)hn;
    }
    {
      const int m = mb1 + v;
      const float r = sigmoidf_(ar1[v] + brb);
      const float z = sigmoidf_(az1[v] + bzb);
      const float nn = tanhf(ai1[v] + bni + r * (ah1[v] + bnh));
      const float hn = (1.0f - z) * nn + z * hf[(long)m * HH + n];
      hof[(long)m * HH + n] = hn;
      hob[(long)m * HH + n] = (bf16)hn;
    }
  }
}

// ---------------------------------------------------------------------------
// Decoder output projections: blocks 0..15 -> t1 = tanh(h@ow_W^T + ow_b),
// blocks 16..23 -> y = h@out_W^T + out_b written straight into d_out[:,t,:].
// ---------------------------------------------------------------------------
__global__ __launch_bounds__(256) void dec_out_kernel(
    const bf16* __restrict__ hbf,
    const bf16* __restrict__ owW, const float* __restrict__ owb,
    const bf16* __restrict__ outW, const float* __restrict__ outb,
    float* __restrict__ t1, float* __restrict__ y, long ldy)
{
  const int wave = threadIdx.x >> 5, lane = threadIdx.x & 31;
  const int ni = wave & 3, mi = wave >> 2;
  v8f acc0 = {}, acc1 = {};
  if (blockIdx.x < 16) {
    const int nt = blockIdx.x * 64 + ni * 16;
#pragma unroll 2
    for (int k0 = 0; k0 < HH; k0 += 32) {
      v16bf b  = load_b_frag(owW, HH, nt, k0);
      v16bf a0 = load_a_frag(hbf, HH, mi * 16, k0);
      v16bf a1 = load_a_frag(hbf, HH, (mi + 2) * 16, k0);
      acc0 = wmma_bf16(a0, b, acc0);
      acc1 = wmma_bf16(a1, b, acc1);
    }
    const int n = nt + (lane & 15);
    const float bb = owb[n];
    const int ma = mi * 16 + (lane >> 4) * 8;
    const int mb = (mi + 2) * 16 + (lane >> 4) * 8;
#pragma unroll
    for (int v = 0; v < 8; ++v) {
      t1[(long)(ma + v) * HH + n] = tanhf(acc0[v] + bb);
      t1[(long)(mb + v) * HH + n] = tanhf(acc1[v] + bb);
    }
  } else {
    const int nt = (blockIdx.x - 16) * 64 + ni * 16;   // N = 512
#pragma unroll 2
    for (int k0 = 0; k0 < HH; k0 += 32) {
      v16bf b  = load_b_frag(outW, HH, nt, k0);
      v16bf a0 = load_a_frag(hbf, HH, mi * 16, k0);
      v16bf a1 = load_a_frag(hbf, HH, (mi + 2) * 16, k0);
      acc0 = wmma_bf16(a0, b, acc0);
      acc1 = wmma_bf16(a1, b, acc1);
    }
    const int n = nt + (lane & 15);
    const float bb = outb[n];
    const int ma = mi * 16 + (lane >> 4) * 8;
    const int mb = (mi + 2) * 16 + (lane >> 4) * 8;
#pragma unroll
    for (int v = 0; v < 8; ++v) {
      y[(long)(ma + v) * ldy + n] = acc0[v] + bb;
      y[(long)(mb + v) * ldy + n] = acc1[v] + bb;
    }
  }
}

// ---------------------------------------------------------------------------
// Local attention (method='general') on precomputed enc_w_full; one block per
// batch row.  sig -> pt/center -> 13 scores -> gauss*softmax -> context c.
// ---------------------------------------------------------------------------
__device__ __forceinline__ float block_reduce_sum(float v, float* sh) {
  const int tid = threadIdx.x;
  sh[tid] = v;
  __syncthreads();
  for (int s = 128; s > 0; s >>= 1) {
    if (tid < s) sh[tid] += sh[tid + s];
    __syncthreads();
  }
  const float r = sh[0];
  __syncthreads();
  return r;
}

__global__ __launch_bounds__(256) void attn_kernel(
    const float* __restrict__ hf, const float* __restrict__ t1,
    const float* __restrict__ ovW, const float* __restrict__ ovb,
    const float* __restrict__ encw,
    float* __restrict__ cf, bf16* __restrict__ cb)
{
  const int b = blockIdx.x, tid = threadIdx.x;
  __shared__ float sh[256];
  __shared__ float s_alpha[WINN];
  __shared__ float s_pt;
  __shared__ int s_center;

  float acc = 0.0f;
  for (int j = tid; j < HH; j += 256) acc += t1[(long)b * HH + j] * ovW[j];
  const float sig = block_reduce_sum(acc, sh);
  if (tid == 0) {
    const float pt = (float)(LSRC - 2 * DD - 1) * sigmoidf_(sig + ovb[0]) + (float)DD;
    s_pt = pt;
    s_center = (int)floorf(pt);
  }
  __syncthreads();
  const float pt = s_pt;
  const int center = s_center;
  const float* hrow = hf + (long)b * HH;
  const float* ewb = encw + (long)b * LSRC * HH;

  for (int w = 0; w < WINN; ++w) {
    const float* er = ewb + (long)(center - DD + w) * HH;
    float p = 0.0f;
    for (int j = tid; j < HH; j += 256) p += er[j] * hrow[j];
    const float ah = block_reduce_sum(p, sh);
    if (tid == 0) {
      const float d = (float)(center - DD + w) - pt;
      s_alpha[w] = ah * __expf(-(d * d) * (1.0f / (2.0f * 3.0f * 3.0f)));  // sigma = D/2 = 3
    }
    __syncthreads();
  }
  if (tid == 0) {
    float mx = s_alpha[0];
    for (int w = 1; w < WINN; ++w) mx = fmaxf(mx, s_alpha[w]);
    float sum = 0.0f;
    for (int w = 0; w < WINN; ++w) { const float e = __expf(s_alpha[w] - mx); s_alpha[w] = e; sum += e; }
    const float inv = 1.0f / sum;
    for (int w = 0; w < WINN; ++w) s_alpha[w] *= inv;
  }
  __syncthreads();
  for (int j = tid; j < HH; j += 256) {
    float c = 0.0f;
#pragma unroll
    for (int w = 0; w < WINN; ++w) c += s_alpha[w] * ewb[(long)(center - DD + w) * HH + j];
    cf[(long)b * HH + j] = c;
    cb[(long)b * HH + j] = (bf16)c;
  }
}

// ---------------------------------------------------------------------------
extern "C" void kernel_launch(void* const* d_in, const int* in_sizes, int n_in,
                              void* d_out, int out_size, void* d_ws, size_t ws_size,
                              hipStream_t stream)
{
  (void)in_sizes; (void)n_in; (void)out_size; (void)ws_size;
  const float* src    = (const float*)d_in[0];
  const float* trg    = (const float*)d_in[1];
  const float* encWih = (const float*)d_in[2];
  const float* encWhh = (const float*)d_in[3];
  const float* encbih = (const float*)d_in[4];
  const float* encbhh = (const float*)d_in[5];
  const float* decWih = (const float*)d_in[6];
  const float* decWhh = (const float*)d_in[7];
  const float* decbih = (const float*)d_in[8];
  const float* decbhh = (const float*)d_in[9];
  const float* outW   = (const float*)d_in[10];
  const float* outb   = (const float*)d_in[11];
  const float* owW    = (const float*)d_in[12];
  const float* owb    = (const float*)d_in[13];
  const float* ovW    = (const float*)d_in[14];
  const float* ovb    = (const float*)d_in[15];
  const float* attnW  = (const float*)d_in[16];
  const float* attnb  = (const float*)d_in[17];
  float* out = (float*)d_out;

  char* ws = (char*)d_ws;
  size_t off = 0;
  auto alloc = [&](size_t bytes) -> void* {
    void* p = ws + off;
    off = (off + bytes + 255) & ~(size_t)255;
    return p;
  };
  float* encw  = (float*)alloc((size_t)B_ * LSRC * HH * 4);     // enc_out @ attn_W^T + b
  bf16*  eob   = (bf16*)alloc((size_t)B_ * LSRC * HH * 2);      // enc_out bf16
  bf16*  srcb  = (bf16*)alloc((size_t)B_ * LSRC * DIMM * 2);
  bf16*  dxb   = (bf16*)alloc((size_t)B_ * TTRG * DIMM * 2);    // shifted trg bf16
  bf16*  eWihb = (bf16*)alloc((size_t)3 * HH * DIMM * 2);
  bf16*  eWhhb = (bf16*)alloc((size_t)3 * HH * HH * 2);
  bf16*  dWihb = (bf16*)alloc((size_t)3 * HH * (DIMM + HH) * 2);
  bf16*  dWhhb = (bf16*)alloc((size_t)3 * HH * HH * 2);
  bf16*  aWb   = (bf16*)alloc((size_t)HH * HH * 2);
  bf16*  owWb  = (bf16*)alloc((size_t)HH * HH * 2);
  bf16*  outWb = (bf16*)alloc((size_t)DIMM * HH * 2);
  float* ehf[2] = { (float*)alloc((size_t)B_ * HH * 4), (float*)alloc((size_t)B_ * HH * 4) };
  bf16*  ehb[2] = { (bf16*)alloc((size_t)B_ * HH * 2), (bf16*)alloc((size_t)B_ * HH * 2) };
  float* dhf[2] = { (float*)alloc((size_t)B_ * HH * 4), (float*)alloc((size_t)B_ * HH * 4) };
  bf16*  dhb[2] = { (bf16*)alloc((size_t)B_ * HH * 2), (bf16*)alloc((size_t)B_ * HH * 2) };
  float* dcf[2] = { (float*)alloc((size_t)B_ * HH * 4), (float*)alloc((size_t)B_ * HH * 4) };
  bf16*  dcb[2] = { (bf16*)alloc((size_t)B_ * HH * 2), (bf16*)alloc((size_t)B_ * HH * 2) };
  float* t1 = (float*)alloc((size_t)B_ * HH * 4);

  // --- precision conversions + initial state ---
  cvt_bf16_kernel<<<1024, 256, 0, stream>>>(src, srcb, (long)B_ * LSRC * DIMM);
  shift_trg_kernel<<<512, 256, 0, stream>>>(trg, dxb);
  cvt_bf16_kernel<<<512, 256, 0, stream>>>(encWih, eWihb, (long)3 * HH * DIMM);
  cvt_bf16_kernel<<<512, 256, 0, stream>>>(encWhh, eWhhb, (long)3 * HH * HH);
  cvt_bf16_kernel<<<512, 256, 0, stream>>>(decWih, dWihb, (long)3 * HH * (DIMM + HH));
  cvt_bf16_kernel<<<512, 256, 0, stream>>>(decWhh, dWhhb, (long)3 * HH * HH);
  cvt_bf16_kernel<<<256, 256, 0, stream>>>(attnW, aWb, (long)HH * HH);
  cvt_bf16_kernel<<<256, 256, 0, stream>>>(owW, owWb, (long)HH * HH);
  cvt_bf16_kernel<<<256, 256, 0, stream>>>(outW, outWb, (long)DIMM * HH);
  zero_f32_kernel<<<64, 256, 0, stream>>>(ehf[0], (long)B_ * HH);
  zero_f32_kernel<<<64, 256, 0, stream>>>((float*)ehb[0], (long)B_ * HH / 2);
  zero_f32_kernel<<<64, 256, 0, stream>>>(dcf[0], (long)B_ * HH);
  zero_f32_kernel<<<64, 256, 0, stream>>>((float*)dcb[0], (long)B_ * HH / 2);

  // --- encoder scan: 500 sequential fused GRU steps ---
  for (int t = 0; t < LSRC; ++t) {
    const int ip = t & 1, op = ip ^ 1;
    enc_step_kernel<<<HH / 64, 256, 0, stream>>>(
        srcb + (long)t * DIMM, (long)LSRC * DIMM,
        ehb[ip], ehf[ip], eWihb, eWhhb, encbih, encbhh,
        ehf[op], ehb[op], eob + (long)t * HH, (long)LSRC * HH);
  }
  float* hTf = ehf[0];  // final parity: ((LSRC-1)&1)^1 == 0
  bf16*  hTb = ehb[0];

  // --- attention projection once for all positions (big parallel WMMA GEMM) ---
  {
    dim3 g(HH / 64, (B_ * LSRC) / 64);  // (16, 500)
    gemm_bias_kernel<<<g, 256, 0, stream>>>(eob, aWb, attnb, encw, HH, HH);
  }

  // --- decoder scan: GRU step -> projections -> attention/context ---
  for (int t = 0; t < TTRG; ++t) {
    const float* hif = (t == 0) ? hTf : dhf[t & 1];
    const bf16*  hib = (t == 0) ? hTb : dhb[t & 1];
    float* hof = dhf[(t + 1) & 1];
    bf16*  hob = dhb[(t + 1) & 1];
    const float* cif_unused = dcf[t & 1]; (void)cif_unused;
    const bf16*  cib = dcb[t & 1];
    float* cof = dcf[(t + 1) & 1];
    bf16*  cob = dcb[(t + 1) & 1];

    dec_step_kernel<<<HH / 64, 256, 0, stream>>>(
        dxb + (long)t * DIMM, (long)TTRG * DIMM,
        cib, hib, hif, dWihb, dWhhb, decbih, decbhh, hof, hob);

    dec_out_kernel<<<24, 256, 0, stream>>>(
        hob, owWb, owb, outWb, outb, t1, out + (long)t * DIMM, (long)TTRG * DIMM);

    attn_kernel<<<B_, 256, 0, stream>>>(hof, t1, ovW, ovb, encw, cof, cob);
  }
}
